// RNNBase_28887950033081
// MI455X (gfx1250) — 8-bit matrix operand order repaired and checked by reading, compiled, not run on hardware
//
#include <hip/hip_runtime.h>
#include <stdint.h>

typedef int v8i __attribute__((ext_vector_type(8)));

constexpr int T_ = 512;
constexpr int B_ = 64;
constexpr int I_ = 1024;
constexpr int H_ = 1024;
constexpr int G_ = 4 * H_;

__device__ inline v8i load_a_frag(const int8_t* __restrict__ p, int half) {
  const int off = 16 * half;
  const int4 d0 = *reinterpret_cast<const int4*>(p + off);
  const int4 d1 = *reinterpret_cast<const int4*>(p + 32 + off);
  v8i a;
  a[0] = d0.x; a[1] = d0.y; a[2] = d0.z; a[3] = d0.w;
  a[4] = d1.x; a[5] = d1.y; a[6] = d1.z; a[7] = d1.w;
  return a;
}

__device__ inline v8i load_b_frag(const int8_t* __restrict__ p, int half) {
  const int off = 16 * half;
  const int4 d0 = *reinterpret_cast<const int4*>(p + off);
  const int4 d1 = *reinterpret_cast<const int4*>(p + 32 + off);
  v8i b;
  b[0] = d0.x; b[1] = d0.y; b[2] = d0.z; b[3] = d0.w;
  b[4] = d1.x; b[5] = d1.y; b[6] = d1.z; b[7] = d1.w;
  return b;
}

__device__ inline float sigmoidf_(float x) { return 1.0f / (1.0f + expf(-x)); }

__global__ __launch_bounds__(256) void zero_f32_kernel(float* p, int n) {
  for (int i = blockIdx.x * blockDim.x + threadIdx.x; i < n;
       i += gridDim.x * blockDim.x)
    p[i] = 0.0f;
}

__global__ __launch_bounds__(256) void pack_i8_kernel(const int* __restrict__ src,
                                                      int8_t* __restrict__ dst, int n) {
  for (int i = blockIdx.x * blockDim.x + threadIdx.x; i < n;
       i += gridDim.x * blockDim.x)
    dst[i] = (int8_t)src[i];
}

__global__ __launch_bounds__(256) void quant_x_kernel(const float* __restrict__ x,
                                                      int8_t* __restrict__ xq,
                                                      float* __restrict__ sx) {
  const int t = blockIdx.x;
  const int n = B_ * I_;
  const float* src = x + (size_t)t * n;
  int8_t* dst = xq + (size_t)t * n;
  __shared__ float red[256];
  float m = 0.0f;
  for (int i = threadIdx.x; i < n; i += 256) m = fmaxf(m, fabsf(src[i]));
  red[threadIdx.x] = m;
  __syncthreads();
  for (int s = 128; s > 0; s >>= 1) {
    if (threadIdx.x < s) red[threadIdx.x] = fmaxf(red[threadIdx.x], red[threadIdx.x + s]);
    __syncthreads();
  }
  const float scale = fmaxf(red[0], 1e-8f) / 127.0f;
  if (threadIdx.x == 0) sx[t] = scale;
  const float inv = 1.0f / scale;
  for (int i = threadIdx.x; i < n; i += 256) {
    float q = rintf(src[i] * inv);
    q = fminf(fmaxf(q, -127.0f), 127.0f);
    dst[i] = (int8_t)q;
  }
}

__global__ __launch_bounds__(256) void absmax_kernel(const float* __restrict__ src,
                                                     int n, float* slot) {
  float m = 0.0f;
  for (int i = blockIdx.x * blockDim.x + threadIdx.x; i < n;
       i += gridDim.x * blockDim.x)
    m = fmaxf(m, fabsf(src[i]));
  atomicMax(reinterpret_cast<unsigned int*>(slot), __float_as_uint(m));
}

__global__ __launch_bounds__(256) void quant_h_kernel(const float* __restrict__ src,
                                                      const float* __restrict__ slot,
                                                      int8_t* __restrict__ dst,
                                                      float* __restrict__ scale_out,
                                                      int n) {
  const float scale = fmaxf(slot[0], 1e-8f) / 127.0f;
  if (blockIdx.x == 0 && threadIdx.x == 0) scale_out[0] = scale;
  const float inv = 1.0f / scale;
  for (int i = blockIdx.x * blockDim.x + threadIdx.x; i < n;
       i += gridDim.x * blockDim.x) {
    float q = rintf(src[i] * inv);
    q = fminf(fmaxf(q, -127.0f), 127.0f);
    dst[i] = (int8_t)q;
  }
}

__global__ __launch_bounds__(128) void lstm_step_kernel(
    const int8_t* __restrict__ xq_t,
    const float*  __restrict__ sx_t,
    const int8_t* __restrict__ hq,
    const float*  __restrict__ sh,
    const int8_t* __restrict__ w_ih8,
    const int8_t* __restrict__ w_hh8,
    const float*  __restrict__ sw_ih,
    const float*  __restrict__ sw_hh,
    const float*  __restrict__ b_ih,
    const float*  __restrict__ b_hh,
    float* __restrict__ c,
    float* __restrict__ h,
    float* __restrict__ ys_t,
    float* __restrict__ absmax_slot)
{
  const int tid  = threadIdx.x;
  const int wave = tid >> 5;
  const int lane = tid & 31;
  const int n    = lane & 15;
  const int half = lane >> 4;
  const int h0c  = blockIdx.x * 16;
  const int row0 = blockIdx.y * 16;
  const int g0   = wave * H_ + h0c;

  const int8_t* aX = xq_t + (size_t)(row0 + n) * I_;
  const int8_t* aH = hq   + (size_t)(row0 + n) * H_;
  const int8_t* bX = w_ih8 + (size_t)(g0 + n) * I_;
  const int8_t* bH = w_hh8 + (size_t)(g0 + n) * H_;

  v8i accX = {0, 0, 0, 0, 0, 0, 0, 0};
  v8i accH = {0, 0, 0, 0, 0, 0, 0, 0};

#pragma unroll 4
  for (int kk = 0; kk < I_; kk += 64) {
    v8i A = load_a_frag(aX + kk, half);
    v8i Bf = load_b_frag(bX + kk, half);
    accX = __builtin_amdgcn_wmma_i32_16x16x64_iu8(true, A, true, Bf, accX,
                                                  false, false);
  }
#pragma unroll 4
  for (int kk = 0; kk < H_; kk += 64) {
    v8i A = load_a_frag(aH + kk, half);
    v8i Bf = load_b_frag(bH + kk, half);
    accH = __builtin_amdgcn_wmma_i32_16x16x64_iu8(true, A, true, Bf, accH,
                                                  false, false);
  }

  const float sxw = sx_t[0] * sw_ih[0];
  const float shw = sh[0]   * sw_hh[0];
  const float bias = b_ih[g0 + n] + b_hh[g0 + n];

  __shared__ float gl[4][16][16];
#pragma unroll
  for (int r = 0; r < 8; ++r) {
    const int m = r + 8 * half;
    gl[wave][m][n] = (float)accX[r] * sxw + (float)accH[r] * shw + bias;
  }
  __syncthreads();

  for (int e = tid; e < 256; e += 128) {
    const int m  = e >> 4;
    const int nn = e & 15;
    const float iv = sigmoidf_(gl[0][m][nn]);
    const float fv = sigmoidf_(gl[1][m][nn]);
    const float gv = tanhf(gl[2][m][nn]);
    const float ov = sigmoidf_(gl[3][m][nn]);
    const size_t idx = (size_t)(row0 + m) * H_ + (h0c + nn);
    const float cn = fv * c[idx] + iv * gv;
    const float hn = ov * tanhf(cn);
    c[idx]   = cn;
    h[idx]   = hn;
    ys_t[idx] = hn;
    atomicMax(reinterpret_cast<unsigned int*>(absmax_slot),
              __float_as_uint(fabsf(hn)));
  }
}

extern "C" void kernel_launch(void* const* d_in, const int* in_sizes, int n_in,
                              void* d_out, int out_size, void* d_ws, size_t ws_size,
                              hipStream_t stream) {
  (void)in_sizes; (void)n_in; (void)out_size; (void)ws_size;

  const float* x       = (const float*)d_in[0];
  const float* h0      = (const float*)d_in[1];
  const float* c0      = (const float*)d_in[2];
  const int*   w_ih_q  = (const int*)d_in[3];
  const int*   w_hh_q  = (const int*)d_in[4];
  const float* sw_ih   = (const float*)d_in[5];
  const float* sw_hh   = (const float*)d_in[6];
  const float* b_ih    = (const float*)d_in[7];
  const float* b_hh    = (const float*)d_in[8];
  float* out = (float*)d_out;

  uint8_t* ws = (uint8_t*)d_ws;
  size_t off = 0;
  auto carve = [&](size_t bytes) {
    uint8_t* p = ws + off;
    off += (bytes + 255) & ~(size_t)255;
    return p;
  };
  int8_t* w_ih8 = (int8_t*)carve((size_t)G_ * I_);
  int8_t* w_hh8 = (int8_t*)carve((size_t)G_ * H_);
  int8_t* xq    = (int8_t*)carve((size_t)T_ * B_ * I_);
  int8_t* hq    = (int8_t*)carve((size_t)B_ * H_);
  float*  cbuf  = (float*)carve((size_t)B_ * H_ * 4);
  float*  hbuf  = (float*)carve((size_t)B_ * H_ * 4);
  float*  sx    = (float*)carve((size_t)T_ * 4);
  float*  shs   = (float*)carve(256);
  float*  amax  = (float*)carve((size_t)(T_ + 1) * 4);

  zero_f32_kernel<<<4, 256, 0, stream>>>(amax, T_ + 1);
  pack_i8_kernel<<<2048, 256, 0, stream>>>(w_ih_q, w_ih8, G_ * I_);
  pack_i8_kernel<<<2048, 256, 0, stream>>>(w_hh_q, w_hh8, G_ * H_);
  quant_x_kernel<<<T_, 256, 0, stream>>>(x, xq, sx);

  hipMemcpyAsync(cbuf, c0, (size_t)B_ * H_ * 4, hipMemcpyDeviceToDevice, stream);
  hipMemcpyAsync(hbuf, h0, (size_t)B_ * H_ * 4, hipMemcpyDeviceToDevice, stream);

  absmax_kernel<<<64, 256, 0, stream>>>(hbuf, B_ * H_, amax + T_);
  quant_h_kernel<<<64, 256, 0, stream>>>(hbuf, amax + T_, hq, shs, B_ * H_);

  dim3 sgrid(H_ / 16, B_ / 16);
  for (int t = 0; t < T_; ++t) {
    lstm_step_kernel<<<sgrid, 128, 0, stream>>>(
        xq + (size_t)t * B_ * I_, sx + t, hq, shs,
        w_ih8, w_hh8, sw_ih, sw_hh, b_ih, b_hh,
        cbuf, hbuf, out + (size_t)t * B_ * H_, amax + t);
    quant_h_kernel<<<64, 256, 0, stream>>>(hbuf, amax + t, hq, shs, B_ * H_);
  }

  float* tail = out + (size_t)T_ * B_ * H_;
  hipMemcpyAsync(tail, hbuf, (size_t)B_ * H_ * 4, hipMemcpyDeviceToDevice, stream);
  hipMemcpyAsync(tail + (size_t)B_ * H_, cbuf, (size_t)B_ * H_ * 4,
                 hipMemcpyDeviceToDevice, stream);
}
